// MultiHeadNeighborhoodAttention_6923487281750
// MI455X (gfx1250) — matrix operand order repaired and checked by reading, not compiled, not run on hardware
//
#include <hip/hip_runtime.h>
#include <hip/hip_bf16.h>

// ---------------------------------------------------------------------------
// MultiHeadNeighborhoodAttention on MI455X (gfx1250, wave32, WMMA)
//
// B=16384, K=64, D=256, H=2, HD=128.
// Algebraic refactor: scores = (q @ Wk) . nb  (exact reassociation) removes
// the 137-GFLOP k-projection; op is HBM-bound on the 1.07 GB neighbor stream.
//
// All matrix math: v_wmma_f32_16x16x32_bf16 (bf16 in, f32 accumulate).
// All WMMA operands are PRE-PACKED into this file's contiguous-k (k = 16h + s) register
// layouts, so each tile fetch is a single 32-byte vector load per lane
// (global_load_b128 / ds_load_b128) instead of 16 scalar u16 loads.
//
//   A-tile (16x32, 16-bit): elem(m,k32) -> lane = m + 16*((k32>>4)&1),
//                                          slot = k32 & 15
//   B-tile (32x16, 16-bit): elem(k32,n) -> lane = n + 16*((k32>>4)&1),
//                                          slot = k32 & 15
//   C-tile (16x16, f32):    elem(m,n)   -> lane = n + 16*(m>=8), vgpr = m&7
// Packed tile = 32 lanes x 16 bf16 = 1 KB, lane-contiguous.
// ---------------------------------------------------------------------------

typedef __attribute__((ext_vector_type(16))) __bf16 v16bf;
typedef __attribute__((ext_vector_type(8)))  float  v8f;

#define NB   16384   // batch
#define ND   256     // model dim
#define NH   2       // heads
#define NHD  128     // head dim
#define NK   64      // neighbors
#define TILE_ELEMS 512  // 32 lanes * 16 elems

__device__ __forceinline__ v8f wmma_bf16(v16bf a, v16bf b, v8f c) {
  return __builtin_amdgcn_wmma_f32_16x16x32_bf16(false, a, false, b, (short)0, c,
                                                 false, false);
}

// flat offset of element (m, k32) inside a packed A tile
__device__ __forceinline__ int a_pos(int m, int k32) {
  int lane = m + ((k32 >> 4) & 1) * 16;
  int slot = k32 & 15;
  return lane * 16 + slot;
}

// one 32-byte vector load per lane fetches a whole operand tile
__device__ __forceinline__ v16bf load_tile(const __bf16* base, int tile, int lane) {
  return *(const v16bf*)(base + (size_t)tile * TILE_ELEMS + lane * 16);
}

// ---------------------------------------------------------------------------
// Kernel 0: weights -> bf16, packed in B-register tile order [kt][nt][lane][16].
//   BQp : B[k=d][n=o]   from Wq_flat[o][d]   (256x256,  kt=8,  nt=16)
//   BKp : B[k=e][n=d]   from Wk[h][e][d]     (2x128x256, per-head kt=4, nt=16)
//   BOp : B[k=j][n=o]   from Wout[o][j]      (512x256,  kt=16, nt=16)
//   BGp : B[k=j][n=o]   from Wgate[o][j]     (512x256,  kt=16, nt=16)
// ---------------------------------------------------------------------------
__global__ __launch_bounds__(256) void prep_kernel(
    const float* __restrict__ Wq, const float* __restrict__ Wk,
    const float* __restrict__ Wout, const float* __restrict__ Wgate,
    __bf16* __restrict__ BQp, __bf16* __restrict__ BKp,
    __bf16* __restrict__ BOp, __bf16* __restrict__ BGp) {
  int i = blockIdx.x * 256 + threadIdx.x;  // packed element index
  int slot = i & 15, lane = (i >> 4) & 31, t = i >> 9;
  int nt = t & 15, kt = t >> 4;
  int n = nt * 16 + (lane & 15);
  int k = kt * 32 + ((lane >> 4) & 1) * 16 + slot;
  if (i < 256 * 256) {
    BQp[i] = (__bf16)Wq[(size_t)n * 256 + k];  // Wq_flat[o=n][d=k]
    int h = kt >> 2;
    int e = (kt & 3) * 32 + ((lane >> 4) & 1) * 16 + slot;
    BKp[i] = (__bf16)Wk[((size_t)h * NHD + e) * ND + n];
  }
  if (i < 512 * 256) {
    BOp[i] = (__bf16)Wout[(size_t)n * 512 + k];
    BGp[i] = (__bf16)Wgate[(size_t)n * 512 + k];
  }
}

// ---------------------------------------------------------------------------
// Kernel 1: per 16-row tile:  q = center @ Wq^T ; p_h = q_h @ Wk[h].
// Writes P bf16 row-major [B][512] (h*256 + d).
// ---------------------------------------------------------------------------
__global__ __launch_bounds__(256) void qp_kernel(const float* __restrict__ center,
                                                 const __bf16* __restrict__ BQp,
                                                 const __bf16* __restrict__ BKp,
                                                 __bf16* __restrict__ P) {
  __shared__ __align__(32) __bf16 sCA[8 * TILE_ELEMS];  // center, A-packed (8 KB)
  __shared__ __align__(32) __bf16 sQA[8 * TILE_ELEMS];  // q, A-packed (8 KB)
  const int row0 = blockIdx.x * 16;
  const int tid  = threadIdx.x;
  const int wave = tid >> 5, lane = tid & 31;

  // stream center tile (f32, b128 loads) -> bf16 A-packed LDS
  const float4* c4 = (const float4*)(center + (size_t)row0 * ND);
  for (int i4 = tid; i4 < 16 * ND / 4; i4 += 256) {
    float4 v = c4[i4];
    int m = i4 >> 6, k0 = (i4 & 63) * 4;
    int p = (k0 >> 5) * TILE_ELEMS + a_pos(m, k0 & 31);
    sCA[p] = (__bf16)v.x; sCA[p + 1] = (__bf16)v.y;
    sCA[p + 2] = (__bf16)v.z; sCA[p + 3] = (__bf16)v.w;
  }
  __syncthreads();

  // GEMM1: q(16x256) = center @ BQ ; wave w owns nt = 2w, 2w+1
#pragma unroll
  for (int t = 0; t < 2; ++t) {
    int nt = wave * 2 + t;
    v8f c = {};
#pragma unroll
    for (int kt = 0; kt < 8; ++kt)
      c = wmma_bf16(load_tile(sCA, kt, lane), load_tile(BQp, kt * 16 + nt, lane), c);
    // scatter C into A-packed sQA (n fixed per lane, m varies with vgpr)
    int n = nt * 16 + (lane & 15), mb = (lane >> 4) * 8;
    int k32 = n & 31, ktg = n >> 5;
    int slot = k32 & 15;
    int lbase = ((k32 >> 4) & 1) * 16;
#pragma unroll
    for (int r = 0; r < 8; ++r)
      sQA[ktg * TILE_ELEMS + (lbase + mb + r) * 16 + slot] = (__bf16)c[r];
  }
  __syncthreads();

  // GEMM2: p_h(16x256) = q[:, h*128:(h+1)*128] @ BK[h] ; 32 C-tiles over 8 waves
#pragma unroll
  for (int t = 0; t < 4; ++t) {
    int ct = wave * 4 + t;
    int h = ct >> 4, nt = ct & 15;
    v8f c = {};
#pragma unroll
    for (int kt = 0; kt < 4; ++kt)
      c = wmma_bf16(load_tile(sQA, h * 4 + kt, lane),
                    load_tile(BKp, (h * 4 + kt) * 16 + nt, lane), c);
    int n = h * ND + nt * 16 + (lane & 15), mb = (lane >> 4) * 8;
#pragma unroll
    for (int r = 0; r < 8; ++r)
      P[(size_t)(row0 + mb + r) * (NH * ND) + n] = (__bf16)c[r];
  }
}

// ---------------------------------------------------------------------------
// Kernel 2: attention, one block per batch row. Streams neighbor[b] (the
// 1.07 GB HBM-bound part) once; stages it in LDS in BOTH A-layout (for the
// score GEMM, NB is the M=64 operand) and B-layout (for the ctx GEMM, NB is
// the K=64 operand). scores and ctx are both WMMA.
// ---------------------------------------------------------------------------
__global__ __launch_bounds__(256) void attn_kernel(
    const float* __restrict__ neighbor, const float* __restrict__ nw,
    const int* __restrict__ mask, const __bf16* __restrict__ P,
    __bf16* __restrict__ Ctx) {
  __shared__ __align__(32) __bf16 sNBA[32 * TILE_ELEMS];  // NB A-packed (32 KB)
  __shared__ __align__(32) __bf16 sNBB[32 * TILE_ELEMS];  // NB B-packed (32 KB)
  __shared__ __align__(32) __bf16 sPB[8 * TILE_ELEMS];    // p  B-packed (8 KB)
  __shared__ __align__(32) __bf16 sAttnA[2 * TILE_ELEMS]; // attn A-packed (2 KB)
  __shared__ float sScore[NH * NK];
  __shared__ float sRed[4];

  const int b   = blockIdx.x;
  const int tid = threadIdx.x;
  const int wave = tid >> 5, lane = tid & 31;

  // stream neighbor tile: f32 b128 loads -> bf16, packed twice
  const float4* nb4 = (const float4*)(neighbor + (size_t)b * NK * ND);
  for (int i4 = tid; i4 < NK * ND / 4; i4 += 256) {
    float4 v = nb4[i4];
    int kk = i4 >> 6, d0 = (i4 & 63) * 4;
    __bf16 x0 = (__bf16)v.x, x1 = (__bf16)v.y, x2 = (__bf16)v.z, x3 = (__bf16)v.w;
    // A-layout: (m=kk, k=d)
    int pA = ((kk >> 4) * 8 + (d0 >> 5)) * TILE_ELEMS + a_pos(kk & 15, d0 & 31);
    sNBA[pA] = x0; sNBA[pA + 1] = x1; sNBA[pA + 2] = x2; sNBA[pA + 3] = x3;
    // B-layout: (k=kk, n=d); 4 consecutive d -> 4 consecutive lanes, same slot
    int pB = ((kk >> 5) * 16 + (d0 >> 4)) * TILE_ELEMS +
             ((d0 & 15) + ((kk >> 4) & 1) * 16) * 16 + (kk & 15);
    sNBB[pB] = x0; sNBB[pB + 16] = x1; sNBB[pB + 32] = x2; sNBB[pB + 48] = x3;
  }
  // p[b] as B matrix [K=256 d][N=16], cols 0,1 = heads, rest zero
  for (int i = tid; i < 8 * TILE_ELEMS; i += 256) {
    int slot = i & 15, ln = (i >> 4) & 31, kt = i >> 9;
    int n = ln & 15;
    int d = kt * 32 + ((ln >> 4) & 1) * 16 + slot;
    sPB[i] = (n < NH) ? P[(size_t)b * (NH * ND) + n * ND + d] : (__bf16)0.0f;
  }
  for (int i = tid; i < 2 * TILE_ELEMS; i += 256) sAttnA[i] = (__bf16)0.0f;
  __syncthreads();

  // score GEMM: NB(64x256) @ pB(256x16): 4 M-tiles (waves 0..3), Kloop=8
  if (wave < 4) {
    v8f c = {};
#pragma unroll
    for (int kt = 0; kt < 8; ++kt)
      c = wmma_bf16(load_tile(sNBA, wave * 8 + kt, lane),
                    load_tile(sPB, kt, lane), c);
    if ((lane & 15) < NH) {  // cols 0,1 of C-tile = heads
      int h = lane & 15, mb = (lane >> 4) * 8;
#pragma unroll
      for (int r = 0; r < 8; ++r) sScore[h * NK + wave * 16 + mb + r] = c[r];
    }
  }
  __syncthreads();

  // scale + weight + mask (f32)
  if (tid < NH * NK) {
    int kk = tid & 63;
    float s = sScore[tid] * 0.08838834764831845f + nw[(size_t)b * NK + kk];
    if (mask[(size_t)b * NK + kk] == 0) s = -1e9f;
    sScore[tid] = s;
  }
  __syncthreads();
  if (tid < NH) {  // 64-wide softmax reduction per head
    float m = -3.0e38f;
    for (int kk = 0; kk < NK; ++kk) m = fmaxf(m, sScore[tid * NK + kk]);
    float sum = 0.f;
    for (int kk = 0; kk < NK; ++kk) sum += __expf(sScore[tid * NK + kk] - m);
    sRed[tid * 2] = m; sRed[tid * 2 + 1] = sum;
  }
  __syncthreads();
  if (tid < NH * NK) {
    int h = tid >> 6, kk = tid & 63;
    float a = __expf(sScore[tid] - sRed[h * 2]) / sRed[h * 2 + 1];
    sAttnA[(kk >> 5) * TILE_ELEMS + a_pos(h, kk & 31)] = (__bf16)a;
  }
  __syncthreads();

  // ctx GEMM: attn(2x64 padded to 16x64) @ NB(64x256): 16 N-tiles, Kloop=2
#pragma unroll
  for (int t = 0; t < 2; ++t) {
    int nt = wave * 2 + t;
    v8f c = {};
#pragma unroll
    for (int kt = 0; kt < 2; ++kt)
      c = wmma_bf16(load_tile(sAttnA, kt, lane),
                    load_tile(sNBB, kt * 16 + nt, lane), c);
    if (lane < 16) {  // rows 0,1 = heads 0,1
      int n = nt * 16 + lane;
      Ctx[(size_t)b * (NH * ND) + n]      = (__bf16)c[0];
      Ctx[(size_t)b * (NH * ND) + ND + n] = (__bf16)c[1];
    }
  }
}

// ---------------------------------------------------------------------------
// Kernel 3: per 16-row tile:
//   proj = ctx(16x512) @ BO ; glog = [center | proj](16x512) @ BG
//   out  = sigmoid(glog)*center + (1-sigmoid)*proj   (f32 blend)
// ---------------------------------------------------------------------------
__global__ __launch_bounds__(256) void out_kernel(const float* __restrict__ center,
                                                  const __bf16* __restrict__ Ctx,
                                                  const __bf16* __restrict__ BOp,
                                                  const __bf16* __restrict__ BGp,
                                                  float* __restrict__ out) {
  __shared__ __align__(32) __bf16 sCtxA[16 * TILE_ELEMS];  // 16 KB (A-packed)
  __shared__ __align__(32) __bf16 sGinA[16 * TILE_ELEMS];  // 16 KB (A-packed)
  __shared__ float sProj[16 * 256];                        // 16 KB
  float* sGl = (float*)sCtxA;  // alias: sCtxA dead after GEMM1 + barrier

  const int row0 = blockIdx.x * 16;
  const int tid  = threadIdx.x;
  const int wave = tid >> 5, lane = tid & 31;

  // ctx tile (bf16 row-major global) -> A-packed LDS
  const __bf16* ct = Ctx + (size_t)row0 * 512;
  for (int i4 = tid; i4 < 16 * 512 / 4; i4 += 256) {
    int m = i4 >> 7, k0 = (i4 & 127) * 4;
    const __bf16* s = ct + m * 512 + k0;
    int p = (k0 >> 5) * TILE_ELEMS + a_pos(m, k0 & 31);
    sCtxA[p] = s[0]; sCtxA[p + 1] = s[1]; sCtxA[p + 2] = s[2]; sCtxA[p + 3] = s[3];
  }
  // center -> gate-input cols 0..255 (A-packed kt 0..7)
  const float4* c4 = (const float4*)(center + (size_t)row0 * ND);
  for (int i4 = tid; i4 < 16 * ND / 4; i4 += 256) {
    float4 v = c4[i4];
    int m = i4 >> 6, k0 = (i4 & 63) * 4;
    int p = (k0 >> 5) * TILE_ELEMS + a_pos(m, k0 & 31);
    sGinA[p] = (__bf16)v.x; sGinA[p + 1] = (__bf16)v.y;
    sGinA[p + 2] = (__bf16)v.z; sGinA[p + 3] = (__bf16)v.w;
  }
  __syncthreads();

  // proj GEMM: 16 N-tiles, Kloop=16
#pragma unroll
  for (int t = 0; t < 2; ++t) {
    int nt = wave * 2 + t;
    v8f c = {};
#pragma unroll
    for (int kt = 0; kt < 16; ++kt)
      c = wmma_bf16(load_tile(sCtxA, kt, lane), load_tile(BOp, kt * 16 + nt, lane), c);
    int n = nt * 16 + (lane & 15), mb = (lane >> 4) * 8;
    int k32 = n & 31;
    int slot = k32 & 15;
    int lbase = ((k32 >> 4) & 1) * 16;
    int ktg = 8 + (n >> 5);  // gate-input col = 256 + n
#pragma unroll
    for (int r = 0; r < 8; ++r) {
      sProj[(mb + r) * 256 + n] = c[r];
      sGinA[ktg * TILE_ELEMS + (lbase + mb + r) * 16 + slot] = (__bf16)c[r];
    }
  }
  __syncthreads();

  // gate GEMM: 16 N-tiles, Kloop=16; results into sGl (aliases sCtxA)
#pragma unroll
  for (int t = 0; t < 2; ++t) {
    int nt = wave * 2 + t;
    v8f c = {};
#pragma unroll
    for (int kt = 0; kt < 16; ++kt)
      c = wmma_bf16(load_tile(sGinA, kt, lane), load_tile(BGp, kt * 16 + nt, lane), c);
    int n = nt * 16 + (lane & 15), mb = (lane >> 4) * 8;
#pragma unroll
    for (int r = 0; r < 8; ++r) sGl[(mb + r) * 256 + n] = c[r];
  }
  __syncthreads();

  for (int i = tid; i < 16 * 256; i += 256) {
    int m = i >> 8, d = i & 255;
    float cen = center[(size_t)(row0 + m) * ND + d];
    float pr  = sProj[i];
    float g   = 1.f / (1.f + __expf(-sGl[i]));
    out[(size_t)(row0 + m) * ND + d] = g * cen + (1.f - g) * pr;
  }
}

// ---------------------------------------------------------------------------
extern "C" void kernel_launch(void* const* d_in, const int* in_sizes, int n_in,
                              void* d_out, int out_size, void* d_ws, size_t ws_size,
                              hipStream_t stream) {
  const float* center   = (const float*)d_in[0];
  const float* neighbor = (const float*)d_in[1];
  const float* nw       = (const float*)d_in[2];
  const int*   mask     = (const int*)d_in[3];
  const float* Wq       = (const float*)d_in[4];
  const float* Wk       = (const float*)d_in[5];
  const float* Wout     = (const float*)d_in[6];
  const float* Wgate    = (const float*)d_in[7];
  float*       out      = (float*)d_out;

  char* ws = (char*)d_ws;
  size_t o = 0;
  __bf16* BQp = (__bf16*)(ws + o); o += (size_t)256 * 256 * 2;    // 128 KB
  __bf16* BKp = (__bf16*)(ws + o); o += (size_t)256 * 256 * 2;    // 128 KB
  __bf16* BOp = (__bf16*)(ws + o); o += (size_t)512 * 256 * 2;    // 256 KB
  __bf16* BGp = (__bf16*)(ws + o); o += (size_t)512 * 256 * 2;    // 256 KB
  __bf16* P   = (__bf16*)(ws + o); o += (size_t)NB * NH * ND * 2; // 16 MB
  __bf16* Cx  = (__bf16*)(ws + o); o += (size_t)NB * NH * ND * 2; // 16 MB

  prep_kernel<<<512, 256, 0, stream>>>(Wq, Wk, Wout, Wgate, BQp, BKp, BOp, BGp);
  qp_kernel<<<NB / 16, 256, 0, stream>>>(center, BQp, BKp, P);
  attn_kernel<<<NB, 256, 0, stream>>>(neighbor, nw, mask, P, Cx);
  out_kernel<<<NB / 16, 256, 0, stream>>>(center, Cx, BOp, BGp, out);
}
